// MultiBanditNet_74715251081488
// MI455X (gfx1250) — matrix operand order repaired and checked by reading, not compiled, not run on hardware
//
#include <hip/hip_runtime.h>
#include <hip/hip_bf16.h>
#include <cstdint>
#include <cstddef>

// Problem dims (fixed by the reference)
#define B_ 2048
#define D_ 512
#define H_ 512
#define O_ 8
#define A_ 256

typedef __attribute__((ext_vector_type(16))) __bf16 v16bf;
typedef __attribute__((ext_vector_type(8)))  float  v8f;
typedef __attribute__((ext_vector_type(4)))  int    v4i;

union Frag16 { v16bf v; v4i q[2]; };   // 32 bytes: one 16x32 (A) or 32x16 (B) bf16 fragment per lane

// ---------------------------------------------------------------------------
// fp32 -> bf16 elementwise convert
// ---------------------------------------------------------------------------
__global__ __launch_bounds__(256) void cvt_f32_bf16(const float* __restrict__ in,
                                                    __hip_bfloat16* __restrict__ out, int n) {
    int i = blockIdx.x * 256 + threadIdx.x;
    if (i < n) out[i] = __float2bfloat16(in[i]);
}

// ---------------------------------------------------------------------------
// Batched transpose + convert: in [G,R,C] fp32 -> out [G,C,R] bf16  (R,C % 32 == 0)
// ---------------------------------------------------------------------------
__global__ __launch_bounds__(256) void transpose_cvt(const float* __restrict__ in,
                                                     __hip_bfloat16* __restrict__ out,
                                                     int R, int C) {
    __shared__ float tile[32][33];
    const int g  = blockIdx.z;
    const float*        src = in  + (size_t)g * R * C;
    __hip_bfloat16*     dst = out + (size_t)g * R * C;
    const int c0 = blockIdx.x * 32, r0 = blockIdx.y * 32;
    const int tx = threadIdx.x & 31, ty = threadIdx.x >> 5;   // 8 rows/pass
    #pragma unroll
    for (int rr = ty; rr < 32; rr += 8)
        tile[rr][tx] = src[(size_t)(r0 + rr) * C + c0 + tx];
    __syncthreads();
    #pragma unroll
    for (int rr = ty; rr < 32; rr += 8)
        dst[(size_t)(c0 + rr) * R + r0 + tx] = __float2bfloat16(tile[tx][rr]);
}

// ---------------------------------------------------------------------------
// Batched WMMA bf16 GEMM over blockIdx.z:
//   C[z][M,N] = A[z][M,K] * Wt[z][N,K]^T + bias[z]
//   A  : bf16 row-major [M,K], batch stride sA   (sA==0 -> shared A)
//   Wt : bf16 row-major [N,K], batch stride sW   (W transposed -> contiguous B frags)
//   relu==1: out = bf16 relu(C)   (activation chaining), batch stride sOut
//   relu==0: out = fp32 C         (logits),              batch stride sOut
// Block: 256 threads (8 waves). Block tile 128x128, wave tile 32x64.
// ---------------------------------------------------------------------------
__global__ __launch_bounds__(256) void gemm_bf16_wmma(
    const __hip_bfloat16* __restrict__ Abase,
    const __hip_bfloat16* __restrict__ Wbase,
    const float* __restrict__ biasBase,
    __hip_bfloat16* __restrict__ outBbase,
    float* __restrict__ outFbase,
    int M, int N, int K, int relu,
    size_t sA, size_t sW, size_t sBias, size_t sOut)
{
    const int z = blockIdx.z;
    const __hip_bfloat16* Abf = Abase + (size_t)z * sA;
    const __hip_bfloat16* Wt  = Wbase + (size_t)z * sW;
    const float* bias = biasBase ? biasBase + (size_t)z * sBias : nullptr;

    const int tid  = threadIdx.x;
    const int lane = tid & 31;
    const int w    = tid >> 5;        // wave 0..7
    const int wm   = w & 3;           // 4 waves along M
    const int wn   = w >> 2;          // 2 waves along N
    const int half = lane >> 4;       // hi/lo lane group
    const int l15  = lane & 15;

    const int m0 = blockIdx.y * 128 + 32 * wm;   // wave's row base (2 x 16)
    const int n0 = blockIdx.x * 128 + 64 * wn;   // wave's col base (4 x 16)

    v8f acc[2][4];
    const v8f vzero = {0.f, 0.f, 0.f, 0.f, 0.f, 0.f, 0.f, 0.f};
    #pragma unroll
    for (int i = 0; i < 2; ++i)
        #pragma unroll
        for (int j = 0; j < 4; ++j) acc[i][j] = vzero;

    for (int k0 = 0; k0 < K; k0 += 32) {
        Frag16 af[2], bf[4];
        // A fragment (16x32 bf16), ISA layout:
        //   lanes 0-15 : row = l15, K = {0..7, 16..23}
        //   lanes 16-31: row = l15, K = {8..15, 24..31}
        #pragma unroll
        for (int i = 0; i < 2; ++i) {
            const __hip_bfloat16* p = Abf + (size_t)(m0 + 16 * i + l15) * K + k0 + 8 * half;
            af[i].q[0] = *reinterpret_cast<const v4i*>(p);
            af[i].q[1] = *reinterpret_cast<const v4i*>(p + 16);
        }
        // B fragment (32x16 bf16), ISA layout:
        //   lane holds column (l15), K = {8*half..+7, 16+8*half..+7} in Wt[N,K]
        #pragma unroll
        for (int j = 0; j < 4; ++j) {
            const __hip_bfloat16* p = Wt + (size_t)(n0 + 16 * j + l15) * K + k0 + 8 * half;
            bf[j].q[0] = *reinterpret_cast<const v4i*>(p);
            bf[j].q[1] = *reinterpret_cast<const v4i*>(p + 16);
        }
        #pragma unroll
        for (int i = 0; i < 2; ++i)
            #pragma unroll
            for (int j = 0; j < 4; ++j)
                acc[i][j] = __builtin_amdgcn_wmma_f32_16x16x32_bf16(
                    false, af[i].v, false, bf[j].v, (short)0, acc[i][j], false, false);
    }

    // Epilogue. C/D layout: VGPR r, lane l -> row = 8*half + r, col = l15
    #pragma unroll
    for (int i = 0; i < 2; ++i) {
        #pragma unroll
        for (int j = 0; j < 4; ++j) {
            const int col = n0 + 16 * j + l15;
            const float bv = bias ? bias[col] : 0.f;
            const int rowBase = m0 + 16 * i + 8 * half;
            union { v8f v; float f[8]; } u;
            u.v = acc[i][j];
            #pragma unroll
            for (int r = 0; r < 8; ++r) {
                float vv = u.f[r] + bv;
                const size_t idx = (size_t)z * sOut + (size_t)(rowBase + r) * N + col;
                if (relu) outBbase[idx] = __float2bfloat16(vv > 0.f ? vv : 0.f);
                else      outFbase[idx] = vv;
            }
        }
    }
}

// ---------------------------------------------------------------------------
// Option head: logits = h[B,H](bf16) @ W2[H,O](f32) + b2 ; softmax + argmax
// One thread per batch row. Writes option_probs and selected_options(as float).
// ---------------------------------------------------------------------------
__global__ __launch_bounds__(256) void option_head_kernel(
    const __hip_bfloat16* __restrict__ h, const float* __restrict__ W2,
    const float* __restrict__ b2, float* __restrict__ out,
    size_t opt_off, size_t selopt_off)
{
    const int b = blockIdx.x * 256 + threadIdx.x;
    if (b >= B_) return;
    float acc[O_];
    #pragma unroll
    for (int o = 0; o < O_; ++o) acc[o] = b2[o];
    const __hip_bfloat16* hr = h + (size_t)b * H_;
    for (int k = 0; k < H_; ++k) {
        const float hv = __bfloat162float(hr[k]);
        #pragma unroll
        for (int o = 0; o < O_; ++o) acc[o] += hv * W2[k * O_ + o];
    }
    float m = acc[0];
    #pragma unroll
    for (int o = 1; o < O_; ++o) m = fmaxf(m, acc[o]);
    float s = 0.f;
    #pragma unroll
    for (int o = 0; o < O_; ++o) { acc[o] = __expf(acc[o] - m); s += acc[o]; }
    const float inv = 1.f / s;
    int best = 0; float bestv = -1.f;
    #pragma unroll
    for (int o = 0; o < O_; ++o) {
        const float p = acc[o] * inv;
        out[opt_off + (size_t)b * O_ + o] = p;
        if (p > bestv) { bestv = p; best = o; }   // first max wins
    }
    out[selopt_off + b] = (float)best;
}

// ---------------------------------------------------------------------------
// Termination head: sigmoid(h[B,H](bf16) @ Wt2[H] + bt2). One thread per row.
// ---------------------------------------------------------------------------
__global__ __launch_bounds__(256) void term_kernel(
    const __hip_bfloat16* __restrict__ h, const float* __restrict__ w,
    const float* __restrict__ bt2, float* __restrict__ out, size_t term_off)
{
    const int b = blockIdx.x * 256 + threadIdx.x;
    if (b >= B_) return;
    float s = bt2[0];
    const __hip_bfloat16* hr = h + (size_t)b * H_;
    for (int k = 0; k < H_; ++k) s += __bfloat162float(hr[k]) * w[k];
    out[term_off + b] = 1.f / (1.f + __expf(-s));
}

// ---------------------------------------------------------------------------
// Row softmax over A=256 + argmax, for ALL O*B rows of logits [O,B,A].
// 8 waves per block; one wave (32 lanes) per row; 8 elems per lane.
// Output layout: action_probs [B,O,A], selected_actions [B,O].
// ---------------------------------------------------------------------------
__global__ __launch_bounds__(256) void softmax_action_kernel(
    const float* __restrict__ logits, float* __restrict__ out,
    size_t act_off, size_t selact_off)
{
    const int wid  = threadIdx.x >> 5;
    const int lane = threadIdx.x & 31;
    const int row  = blockIdx.x * 8 + wid;       // 0 .. O_*B_-1
    const int o    = row / B_;
    const int b    = row - o * B_;
    const float* src = logits + (size_t)row * A_;
    float v[8];
    float m = -3.402823466e+38f;
    #pragma unroll
    for (int t = 0; t < 8; ++t) { v[t] = src[lane + 32 * t]; m = fmaxf(m, v[t]); }
    #pragma unroll
    for (int off = 16; off >= 1; off >>= 1) m = fmaxf(m, __shfl_xor(m, off, 32));
    float s = 0.f;
    #pragma unroll
    for (int t = 0; t < 8; ++t) { v[t] = __expf(v[t] - m); s += v[t]; }
    #pragma unroll
    for (int off = 16; off >= 1; off >>= 1) s += __shfl_xor(s, off, 32);
    const float inv = 1.f / s;
    float* dst = out + act_off + (size_t)b * (O_ * A_) + (size_t)o * A_;
    float bv = -1.f; int bi = A_;
    #pragma unroll
    for (int t = 0; t < 8; ++t) {
        const float p = v[t] * inv;
        const int a = lane + 32 * t;
        dst[a] = p;
        if (p > bv || (p == bv && a < bi)) { bv = p; bi = a; }
    }
    #pragma unroll
    for (int off = 16; off >= 1; off >>= 1) {
        const float ov = __shfl_xor(bv, off, 32);
        const int   oi = __shfl_xor(bi, off, 32);
        if (ov > bv || (ov == bv && oi < bi)) { bv = ov; bi = oi; }
    }
    if (lane == 0) out[selact_off + (size_t)b * O_ + o] = (float)bi;
}

// ---------------------------------------------------------------------------
// Host-side orchestration
// ---------------------------------------------------------------------------
extern "C" void kernel_launch(void* const* d_in, const int* in_sizes, int n_in,
                              void* d_out, int out_size, void* d_ws, size_t ws_size,
                              hipStream_t stream) {
    (void)in_sizes; (void)n_in; (void)out_size; (void)ws_size;
    // Inputs, in setup_inputs() order
    const float* state  = (const float*)d_in[0];
    const float* W_opt1 = (const float*)d_in[1];
    const float* b_opt1 = (const float*)d_in[2];
    const float* W_opt2 = (const float*)d_in[3];
    const float* b_opt2 = (const float*)d_in[4];
    const float* Wa_in  = (const float*)d_in[5];
    const float* ba_in  = (const float*)d_in[6];
    const float* Wa_hid = (const float*)d_in[7];
    const float* ba_hid = (const float*)d_in[8];
    const float* Wa_out = (const float*)d_in[9];
    const float* ba_out = (const float*)d_in[10];
    const float* Wt1    = (const float*)d_in[11];
    const float* bt1    = (const float*)d_in[12];
    const float* Wt2    = (const float*)d_in[13];
    const float* bt2    = (const float*)d_in[14];
    float* out = (float*)d_out;

    // Output offsets (flat f32 elements, in reference return order)
    const size_t OPT_OFF    = 0;
    const size_t ACT_OFF    = (size_t)B_ * O_;                       // 16384
    const size_t TERM_OFF   = ACT_OFF + (size_t)B_ * O_ * A_;        // 4,210,688
    const size_t SELOPT_OFF = TERM_OFF + B_;
    const size_t SELACT_OFF = SELOPT_OFF + B_;

    // Workspace layout (bytes; KiB-granular offsets)
    char* ws = (char*)d_ws;
    __hip_bfloat16* stateBf = (__hip_bfloat16*)(ws);                     // B*D bf16        = 2 MiB
    __hip_bfloat16* Wopt1T  = (__hip_bfloat16*)(ws + (2048u  << 10));    // H*D bf16        = 512 KiB
    __hip_bfloat16* Wt1T    = (__hip_bfloat16*)(ws + (2560u  << 10));    // H*D bf16        = 512 KiB
    __hip_bfloat16* WaInT   = (__hip_bfloat16*)(ws + (3072u  << 10));    // O*H*D bf16      = 4 MiB
    __hip_bfloat16* WaHidT  = (__hip_bfloat16*)(ws + (7168u  << 10));    // O*2*H*H bf16    = 8 MiB
    __hip_bfloat16* WaOutT  = (__hip_bfloat16*)(ws + (15360u << 10));    // O*A*H bf16      = 2 MiB
    __hip_bfloat16* actA    = (__hip_bfloat16*)(ws + (17408u << 10));    // [O,B,H] bf16    = 16 MiB
    __hip_bfloat16* actB    = (__hip_bfloat16*)(ws + (33792u << 10));    // [O,B,H] bf16    = 16 MiB
    float*          logits  = (float*)(ws + (50176u << 10));             // [O,B,A] f32     = 16 MiB

    // ---- Precision conversion + weight transposes (to [N,K] bf16) ----
    {
        const int n = B_ * D_;
        cvt_f32_bf16<<<(n + 255) / 256, 256, 0, stream>>>(state, stateBf, n);
    }
    transpose_cvt<<<dim3(H_ / 32, D_ / 32, 1),      256, 0, stream>>>(W_opt1, Wopt1T, D_, H_);
    transpose_cvt<<<dim3(H_ / 32, D_ / 32, 1),      256, 0, stream>>>(Wt1,    Wt1T,   D_, H_);
    transpose_cvt<<<dim3(H_ / 32, D_ / 32, O_),     256, 0, stream>>>(Wa_in,  WaInT,  D_, H_);
    transpose_cvt<<<dim3(H_ / 32, H_ / 32, 2 * O_), 256, 0, stream>>>(Wa_hid, WaHidT, H_, H_);
    transpose_cvt<<<dim3(A_ / 32, H_ / 32, O_),     256, 0, stream>>>(Wa_out, WaOutT, H_, A_);

    const dim3 blk(256);
    const dim3 gH8(H_ / 128, B_ / 128, O_);   // batched over options, N=512
    const dim3 gA8(A_ / 128, B_ / 128, O_);   // batched over options, N=256
    const dim3 gH1(H_ / 128, B_ / 128, 1);

    const size_t sAct = (size_t)B_ * H_;      // activation batch stride (elements)

    // ---- Per-option MLPs, batched over blockIdx.z (512 WGs/launch) ----
    // layer in : state -> actA
    gemm_bf16_wmma<<<gH8, blk, 0, stream>>>(stateBf, WaInT, ba_in, actA, nullptr,
                                            B_, H_, D_, 1,
                                            0, (size_t)H_ * D_, (size_t)H_, sAct);
    // hid 0 : actA -> actB
    gemm_bf16_wmma<<<gH8, blk, 0, stream>>>(actA, WaHidT, ba_hid, actB, nullptr,
                                            B_, H_, H_, 1,
                                            sAct, (size_t)2 * H_ * H_, (size_t)2 * H_, sAct);
    // hid 1 : actB -> actA   (weights/bias offset by one layer)
    gemm_bf16_wmma<<<gH8, blk, 0, stream>>>(actB, WaHidT + (size_t)H_ * H_, ba_hid + H_,
                                            actA, nullptr,
                                            B_, H_, H_, 1,
                                            sAct, (size_t)2 * H_ * H_, (size_t)2 * H_, sAct);
    // out : actA -> logits (fp32)
    gemm_bf16_wmma<<<gA8, blk, 0, stream>>>(actA, WaOutT, ba_out, nullptr, logits,
                                            B_, A_, H_, 0,
                                            sAct, (size_t)A_ * H_, (size_t)A_, (size_t)B_ * A_);
    // softmax + argmax over all O*B rows
    softmax_action_kernel<<<(O_ * B_) / 8, blk, 0, stream>>>(logits, out, ACT_OFF, SELACT_OFF);

    // ---- Option head ----
    gemm_bf16_wmma<<<gH1, blk, 0, stream>>>(stateBf, Wopt1T, b_opt1, actB, nullptr,
                                            B_, H_, D_, 1, 0, 0, 0, 0);
    option_head_kernel<<<B_ / 256, blk, 0, stream>>>(actB, W_opt2, b_opt2, out, OPT_OFF, SELOPT_OFF);

    // ---- Termination head ----
    gemm_bf16_wmma<<<gH1, blk, 0, stream>>>(stateBf, Wt1T, bt1, actA, nullptr,
                                            B_, H_, D_, 1, 0, 0, 0, 0);
    term_kernel<<<B_ / 256, blk, 0, stream>>>(actA, Wt2, bt2, out, TERM_OFF);
}
